// TGN_46248207843702
// MI455X (gfx1250) — compile-verified, matrix operand order checked by reading, not run on hardware
//
#include <hip/hip_runtime.h>
#include <hip/hip_bf16.h>
#include <math.h>

// ---------------- problem constants (match reference) ----------------
#define NN      200000      // NUM_NODES
#define EE      65536       // edges
#define MEMD    128
#define TD      100
#define EDGED   128
#define MSGD    256         // message dim
#define XK      484         // 2*MEMD + TD + EDGED
#define XKP     512         // padded K for message GEMM (16 chunks of 32)
// LDS strides (halves) padded so 16 rows rotate across all 64 banks
#define XSTR    520         // 512 + 8
#define ASTR    264         // 256 + 8
#define MSTR    136         // 128 + 8

// ---------------- workspace layout (bytes) ----------------
#define OFF_WMSG   0ull                         // 16*16*32*16 halves = 262144 B
#define OFF_WG     262144ull                    // 24*12*32*16 halves = 294912 B
#define OFF_MSUM   557056ull                    // NN*256*4 = 204,800,000 B
#define OFF_CNT    (OFF_MSUM + 204800000ull)    // NN*4
#define OFF_TSM    (OFF_CNT  + 800000ull)       // NN*4
#define ZERO_BYTES 206400000ull                 // msum + cnt + tsm (contiguous)

typedef _Float16 hvec16 __attribute__((ext_vector_type(16)));
typedef _Float16 hvec8  __attribute__((ext_vector_type(8)));
typedef float    fvec8  __attribute__((ext_vector_type(8)));

__device__ __forceinline__ int korder(int lane, int h) {
  // 16-bit WMMA A/B fragment K mapping (ISA 7.12.2)
  return (h & 7) + ((h >> 3) << 4) + ((lane >> 4) << 3);
}

__device__ __forceinline__ hvec16 load_a_frag(const _Float16* row, int k0, int lane) {
  // halves 0..7 -> k = base..base+7 ; halves 8..15 -> k = base+16..base+23
  int base = k0 + ((lane >> 4) << 3);
  hvec8 lo = *(const hvec8*)(row + base);       // ds_load_b128
  hvec8 hi = *(const hvec8*)(row + base + 16);  // ds_load_b128
  hvec16 a;
#pragma unroll
  for (int i = 0; i < 8; ++i) { a[i] = lo[i]; a[8 + i] = hi[i]; }
  return a;
}

__device__ __forceinline__ fvec8 wmma16(hvec16 a, hvec16 b, fvec8 c) {
  return __builtin_amdgcn_wmma_f32_16x16x32_f16(false, a, false, b, (short)0, c,
                                                false, false);
}

__device__ __forceinline__ float sigmoidf_(float x) {
  return 1.0f / (1.0f + __expf(-x));
}

// Guaranteed-native CDNA5 atomics (non-returning; tracked by STOREcnt,
// drained by the implicit wait-idle at s_endpgm; consumer is a later kernel
// on the same stream, so L2-level RMW ordering is sufficient).
__device__ __forceinline__ void atom_add_f32(float* p, float v) {
  asm volatile("global_atomic_add_f32 %0, %1, off" : : "v"(p), "v"(v) : "memory");
}
__device__ __forceinline__ void atom_max_u32(unsigned* p, unsigned v) {
  asm volatile("global_atomic_max_u32 %0, %1, off" : : "v"(p), "v"(v) : "memory");
}

// ---------------- 0) zero scratch (msg_sum, counts, ts-max) ----------------
__global__ void k_zero(uint4* p, long n4) {
  long i = (long)blockIdx.x * blockDim.x + threadIdx.x;
  long stride = (long)gridDim.x * blockDim.x;
  uint4 z = make_uint4(0u, 0u, 0u, 0u);
  for (; i < n4; i += stride) p[i] = z;
}

// ---------------- 1) pre-swizzle W_msg into B fragments -------------------
// frag[nt][kc][lane][h], nt in 0..15 (N=256), kc in 0..15 (K=512 padded)
__global__ void k_swz_msg(const float* __restrict__ Wmsg, _Float16* __restrict__ out) {
  int tid = blockIdx.x * blockDim.x + threadIdx.x;
  if (tid >= 16 * 16 * 32) return;
  int lane = tid & 31, kc = (tid >> 5) & 15, nt = tid >> 9;
  int n = nt * 16 + (lane & 15);
  _Float16* o = out + ((size_t)(nt * 16 + kc) * 32 + lane) * 16;
#pragma unroll
  for (int h = 0; h < 16; ++h) {
    int k = kc * 32 + korder(lane, h);
    float v = (k < XK) ? Wmsg[(size_t)n * XK + k] : 0.0f;
    o[h] = (_Float16)v;
  }
}

// ---------------- 2) pre-swizzle [W_ih | W_hh] (K-concat) -----------------
// frag[nt][kc][lane][h], nt in 0..23 (N=384), kc 0..7 -> W_ih K, kc 8..11 -> W_hh K
__global__ void k_swz_g(const float* __restrict__ Wih, const float* __restrict__ Whh,
                        _Float16* __restrict__ out) {
  int tid = blockIdx.x * blockDim.x + threadIdx.x;
  if (tid >= 24 * 12 * 32) return;
  int lane = tid & 31, kc = (tid / 32) % 12, nt = tid / (32 * 12);
  int n = nt * 16 + (lane & 15);
  _Float16* o = out + ((size_t)(nt * 12 + kc) * 32 + lane) * 16;
#pragma unroll
  for (int h = 0; h < 16; ++h) {
    int kk = korder(lane, h);
    float v;
    if (kc < 8) v = Wih[(size_t)n * 256 + kc * 32 + kk];
    else        v = Whh[(size_t)n * 128 + (kc - 8) * 32 + kk];
    o[h] = (_Float16)v;
  }
}

// ---------------- 3) edge kernel: messages + scatter ----------------------
// 16 edges per block; wave 0 = src-direction, wave 1 = dst-direction.
__global__ void __launch_bounds__(64)
k_edge(const int* __restrict__ src, const int* __restrict__ dst,
       const float* __restrict__ ts, const float* __restrict__ ef,
       const float* __restrict__ mem, const float* __restrict__ last,
       const float* __restrict__ bf, const float* __restrict__ ph,
       const float* __restrict__ bmsg, const _Float16* __restrict__ wfrag,
       float* __restrict__ msum, float* __restrict__ cnt,
       unsigned* __restrict__ tsm) {
  __shared__ _Float16 xs[16][XSTR];   // x_src tile (f16, padded stride)
  __shared__ _Float16 xd[16][XSTR];   // x_dst tile
  __shared__ int sid[16], did[16];
  __shared__ float dts[16], dtd[16];

  int tid = threadIdx.x;
  int e0 = blockIdx.x * 16;

  if (tid < 16) {
    int e = e0 + tid;
    int s = src[e], d = dst[e];
    float t = ts[e];
    sid[tid] = s; did[tid] = d;
    dts[tid] = t - last[s];
    dtd[tid] = t - last[d];
    // counts + timestamp max (ts >= 0 so float bits order as uint)
    atom_add_f32(&cnt[s], 1.0f);
    atom_add_f32(&cnt[d], 1.0f);
    unsigned tb = __float_as_uint(t);
    atom_max_u32(&tsm[s], tb);
    atom_max_u32(&tsm[d], tb);
  }
  __syncthreads();

  // Build both 16x512 f16 input tiles cooperatively (coalesced along k).
  for (int idx = tid; idx < 16 * XKP; idx += 64) {
    int e = idx >> 9, k = idx & (XKP - 1);
    float vs, vd;
    if (k < MEMD) {
      vs = mem[(size_t)did[e] * MEMD + k];
      vd = mem[(size_t)sid[e] * MEMD + k];
    } else if (k < 2 * MEMD) {
      int kk = k - MEMD;
      vs = mem[(size_t)sid[e] * MEMD + kk];
      vd = mem[(size_t)did[e] * MEMD + kk];
    } else if (k < 2 * MEMD + TD) {
      int kk = k - 2 * MEMD;
      float f = bf[kk], p = ph[kk];
      vs = __cosf(dts[e] * f + p);
      vd = __cosf(dtd[e] * f + p);
    } else if (k < XK) {
      float v = ef[(size_t)(e0 + e) * EDGED + (k - 2 * MEMD - TD)];
      vs = v; vd = v;
    } else { vs = 0.0f; vd = 0.0f; }
    xs[e][k] = (_Float16)vs;
    xd[e][k] = (_Float16)vd;
  }
  __syncthreads();

  int w = tid >> 5, lane = tid & 31;
  const _Float16* X = w ? &xd[0][0] : &xs[0][0];
  const int* nid = w ? did : sid;
  const _Float16* Arow = X + (size_t)(lane & 15) * XSTR;
  int mhalf = (lane >> 4) * 8;
  int nodes[8];
#pragma unroll
  for (int vr = 0; vr < 8; ++vr) nodes[vr] = nid[mhalf + vr];

  for (int nt = 0; nt < 16; ++nt) {          // 16 output tiles of 16 cols
    fvec8 acc = (fvec8)(0.0f);
#pragma unroll
    for (int kc = 0; kc < 16; ++kc) {        // K = 512 in chunks of 32
      hvec16 a = load_a_frag(Arow, kc * 32, lane);
      hvec16 b = *(const hvec16*)(wfrag + ((size_t)(nt * 16 + kc) * 32 + lane) * 16);
      acc = wmma16(a, b, acc);
    }
    int n = nt * 16 + (lane & 15);
    float bias = bmsg[n];
#pragma unroll
    for (int vr = 0; vr < 8; ++vr) {
      float v = fmaxf(acc[vr] + bias, 0.0f);          // relu
      atom_add_f32(&msum[(size_t)nodes[vr] * MSGD + n], v);
    }
  }
}

// ---------------- 4) GRU update over all nodes ----------------------------
// 16 nodes per block; 2 waves; wave w owns col-tiles t = w*4 .. w*4+3.
// For column j = t*16 + (lane&15): r-tile t, z-tile t+8, n-tiles t+16 —
// all land in the same lane/VGPR slot, so the gate math is per-lane VALU.
__global__ void __launch_bounds__(64)
k_gru(const float* __restrict__ mem, const float* __restrict__ msum,
      const float* __restrict__ cnt, const _Float16* __restrict__ gfrag,
      const float* __restrict__ bih, const float* __restrict__ bhh,
      float* __restrict__ out) {
  __shared__ _Float16 aggh[16][ASTR];  // agg (f16) K=256
  __shared__ _Float16 memh[16][MSTR];  // mem (f16) K=128
  __shared__ float memf[16][MEMD];     // mem (f32) for blend
  __shared__ float cntl[16];

  int tid = threadIdx.x;
  int base = blockIdx.x * 16;

  if (tid < 16) cntl[tid] = cnt[base + tid];
  for (int idx = tid; idx < 16 * ASTR; idx += 64) {
    int m = idx / ASTR, k = idx % ASTR;
    float v = 0.0f;
    if (k < MSGD) {
      float c = fmaxf(cnt[base + m], 1.0f);
      v = msum[(size_t)(base + m) * MSGD + k] / c;
    }
    aggh[m][k] = (_Float16)v;
  }
  for (int idx = tid; idx < 16 * MSTR; idx += 64) {
    int m = idx / MSTR, k = idx % MSTR;
    float v = (k < MEMD) ? mem[(size_t)(base + m) * MEMD + k] : 0.0f;
    memh[m][k] = (_Float16)v;
    if (k < MEMD) memf[m][k] = v;
  }
  __syncthreads();

  int w = tid >> 5, lane = tid & 31;
  const _Float16* Aagg = &aggh[lane & 15][0];
  const _Float16* Amem = &memh[lane & 15][0];
  int mhalf = (lane >> 4) * 8;

  for (int t = w * 4; t < w * 4 + 4; ++t) {
    fvec8 ar = (fvec8)(0.0f), az = (fvec8)(0.0f);
    fvec8 ain = (fvec8)(0.0f), ahn = (fvec8)(0.0f);
#pragma unroll
    for (int kc = 0; kc < 12; ++kc) {    // r-gate: fused agg+mem K
      hvec16 a = (kc < 8) ? load_a_frag(Aagg, kc * 32, lane)
                          : load_a_frag(Amem, (kc - 8) * 32, lane);
      hvec16 b = *(const hvec16*)(gfrag + ((size_t)(t * 12 + kc) * 32 + lane) * 16);
      ar = wmma16(a, b, ar);
    }
#pragma unroll
    for (int kc = 0; kc < 12; ++kc) {    // z-gate: fused agg+mem K
      hvec16 a = (kc < 8) ? load_a_frag(Aagg, kc * 32, lane)
                          : load_a_frag(Amem, (kc - 8) * 32, lane);
      hvec16 b = *(const hvec16*)(gfrag + ((size_t)((t + 8) * 12 + kc) * 32 + lane) * 16);
      az = wmma16(a, b, az);
    }
#pragma unroll
    for (int kc = 0; kc < 8; ++kc) {     // i_n (agg @ W_ih part)
      hvec16 a = load_a_frag(Aagg, kc * 32, lane);
      hvec16 b = *(const hvec16*)(gfrag + ((size_t)((t + 16) * 12 + kc) * 32 + lane) * 16);
      ain = wmma16(a, b, ain);
    }
#pragma unroll
    for (int kc = 8; kc < 12; ++kc) {    // h_n (mem @ W_hh part) kept separate
      hvec16 a = load_a_frag(Amem, (kc - 8) * 32, lane);
      hvec16 b = *(const hvec16*)(gfrag + ((size_t)((t + 16) * 12 + kc) * 32 + lane) * 16);
      ahn = wmma16(a, b, ahn);
    }

    int col = t * 16 + (lane & 15);      // 0..127
    float br = bih[col]        + bhh[col];
    float bz = bih[MEMD + col] + bhh[MEMD + col];
    float bin = bih[2 * MEMD + col];
    float bhn = bhh[2 * MEMD + col];
#pragma unroll
    for (int vr = 0; vr < 8; ++vr) {
      int m = vr + mhalf;
      float r  = sigmoidf_(ar[vr] + br);
      float z  = sigmoidf_(az[vr] + bz);
      float nn = tanhf(ain[vr] + bin + r * (ahn[vr] + bhn));
      float mv = memf[m][col];
      float h  = (1.0f - z) * nn + z * mv;
      out[(size_t)(base + m) * MEMD + col] = (cntl[m] > 0.0f) ? h : mv;
    }
  }
}

// ---------------- 5) timestamp finalize -----------------------------------
__global__ void k_ts(const float* __restrict__ last, const unsigned* __restrict__ tsm,
                     float* __restrict__ out_last, int n) {
  int i = blockIdx.x * blockDim.x + threadIdx.x;
  if (i < n) out_last[i] = fmaxf(last[i], __uint_as_float(tsm[i]));
}

// ---------------- launcher -------------------------------------------------
extern "C" void kernel_launch(void* const* d_in, const int* in_sizes, int n_in,
                              void* d_out, int out_size, void* d_ws, size_t ws_size,
                              hipStream_t stream) {
  const int*   src  = (const int*)d_in[0];
  const int*   dst  = (const int*)d_in[1];
  const float* ts   = (const float*)d_in[2];
  const float* ef   = (const float*)d_in[3];
  const float* mem  = (const float*)d_in[4];
  const float* last = (const float*)d_in[5];
  const float* bf   = (const float*)d_in[6];
  const float* ph   = (const float*)d_in[7];
  const float* Wmsg = (const float*)d_in[8];
  const float* bmsg = (const float*)d_in[9];
  const float* Wih  = (const float*)d_in[10];
  const float* Whh  = (const float*)d_in[11];
  const float* bih  = (const float*)d_in[12];
  const float* bhh  = (const float*)d_in[13];

  char* ws = (char*)d_ws;
  _Float16* wmsgf = (_Float16*)(ws + OFF_WMSG);
  _Float16* wgf   = (_Float16*)(ws + OFF_WG);
  float*    msum  = (float*)(ws + OFF_MSUM);
  float*    cnt   = (float*)(ws + OFF_CNT);
  unsigned* tsm   = (unsigned*)(ws + OFF_TSM);
  float*    out   = (float*)d_out;

  // zero msg_sum / counts / ts-max (contiguous region)
  k_zero<<<2048, 256, 0, stream>>>((uint4*)(ws + OFF_MSUM), (long)(ZERO_BYTES / 16));
  // pre-swizzle weights into WMMA B-fragment layout
  k_swz_msg<<<(16 * 16 * 32 + 255) / 256, 256, 0, stream>>>(Wmsg, wmsgf);
  k_swz_g<<<(24 * 12 * 32 + 255) / 256, 256, 0, stream>>>(Wih, Whh, wgf);
  // per-edge messages + scatter-add
  k_edge<<<EE / 16, 64, 0, stream>>>(src, dst, ts, ef, mem, last, bf, ph,
                                     bmsg, wmsgf, msum, cnt, tsm);
  // GRU memory update (writes new_memory)
  k_gru<<<NN / 16, 64, 0, stream>>>(mem, msum, cnt, wgf, bih, bhh, out);
  // last-update timestamps (second output, flat-concatenated)
  k_ts<<<(NN + 255) / 256, 256, 0, stream>>>(last, tsm, out + (size_t)NN * MEMD, NN);
}
